// HATS_38431367365105
// MI455X (gfx1250) — matrix operand order repaired and checked by reading, not compiled, not run on hardware
//
#include <hip/hip_runtime.h>
#include <hip/hip_bf16.h>
#include <math.h>

typedef __attribute__((ext_vector_type(16))) __bf16 v16bf;
typedef __attribute__((ext_vector_type(8)))  float  v8f;

#define B_ 4
#define N_ 2000
#define T_ 50
#define F_ 10
#define D_ 64
#define R_ 8
#define K_ 16
#define L_ 3
#define E_ 32

__device__ __forceinline__ float wred32(float v) {
#pragma unroll
  for (int off = 16; off > 0; off >>= 1) v += __shfl_xor(v, off, 32);
  return v;
}
// branch-free native-rate math (v_rcp_f32 / v_exp_f32), saturation-safe
__device__ __forceinline__ float frcp_(float x) { return __builtin_amdgcn_rcpf(x); }
__device__ __forceinline__ float sigmoidf_(float x) { return frcp_(1.0f + __expf(-x)); }
__device__ __forceinline__ float tanhf_(float x) { return 1.0f - 2.0f * frcp_(1.0f + __expf(2.0f * x)); }
__device__ __forceinline__ float leakyf_(float x) { return x >= 0.0f ? x : 0.2f * x; }

// A-fragment K index, contiguous order (k = 16*half + element); B fragments use the same.
__device__ __forceinline__ int aK(int v, int half) {
  return 2 * v + 16 * half;
}

// ---------------------------------------------------------------------------
// Prep: projT bf16 [r][e][d] (transposed so B-frag pairs are contiguous),
// and zero the padding row (index 0) of node_pad for each batch.
// ---------------------------------------------------------------------------
__global__ void prep_kernel(const float* __restrict__ proj_W,
                            __bf16* __restrict__ projT,
                            float* __restrict__ node_pad) {
  int i = blockIdx.x * blockDim.x + threadIdx.x;
  if (i < B_ * D_) {
    int b = i >> 6, d = i & 63;
    node_pad[((size_t)b * (N_ + 1)) * D_ + d] = 0.0f;
  }
  if (i < R_ * D_ * D_) {
    int r = i >> 12;
    int rem = i & 4095;
    int e = rem >> 6;
    int d = rem & 63;
    projT[i] = (__bf16)proj_W[r * 4096 + d * 64 + e];
  }
}

// ---------------------------------------------------------------------------
// GRU: one wave per 16 sequences. Per step, 4 column groups x (3+3+1+2) WMMA.
// Hidden state carried in f32 registers; bf16 copy in LDS for A-fragments.
// All LDS tiles zero-padded so fragment builds are branch-free.
// ---------------------------------------------------------------------------
__global__ __launch_bounds__(32) void gru_kernel(const float* __restrict__ X,
                                                 const float* __restrict__ W,
                                                 const float* __restrict__ U,
                                                 const float* __restrict__ bvec,
                                                 float* __restrict__ node_pad) {
  __shared__ __bf16 sWt[192 * 32];  // [col][k], k>=10 zero-padded
  __shared__ __bf16 sUt[192 * 64];  // [col][k]
  __shared__ __bf16 sHb[16 * 64];   // bf16 copy of h, [row][k]
  __shared__ __bf16 sXb[16 * 32];   // bf16 x_t tile, k>=10 zero-padded

  const int lane = threadIdx.x;
  const int seq0 = blockIdx.x * 16;
  const int half = lane >> 4;
  const int l16  = lane & 15;

  for (int i = lane; i < 192 * 32; i += 32) {
    int col = i >> 5, k = i & 31;
    sWt[i] = (__bf16)((k < F_) ? W[k * 192 + col] : 0.0f);
  }
  for (int i = lane; i < 192 * 64; i += 32) {
    int col = i >> 6, k = i & 63;
    sUt[i] = (__bf16)U[k * 192 + col];
  }
  for (int i = lane; i < 16 * 64; i += 32) sHb[i] = (__bf16)0.0f;
  for (int i = lane; i < 16 * 32; i += 32) sXb[i] = (__bf16)0.0f;

  // Per-lane bias values for this lane's 4 column positions.
  float bz[4], br[4], bxh[4], buh[4];
#pragma unroll
  for (int j = 0; j < 4; ++j) {
    int col = j * 16 + l16;
    bz[j]  = bvec[col] + bvec[192 + col];
    br[j]  = bvec[64 + col] + bvec[192 + 64 + col];
    bxh[j] = bvec[128 + col];
    buh[j] = bvec[192 + 128 + col];
  }
  float hreg[4][8];
#pragma unroll
  for (int j = 0; j < 4; ++j)
#pragma unroll
    for (int i = 0; i < 8; ++i) hreg[j][i] = 0.0f;
  __syncthreads();

  for (int t = 0; t < T_; ++t) {
    // stage x_t tile as bf16 (cols 10..31 stay zero)
    for (int i = lane; i < 16 * F_; i += 32) {
      int row = i / F_, f = i % F_;
      sXb[row * 32 + f] = (__bf16)X[((size_t)(seq0 + row) * T_ + t) * F_ + f];
    }
    __syncthreads();

    // A fragments: branch-free contiguous LDS reads
    v16bf aX, aH0, aH1;
#pragma unroll
    for (int v = 0; v < 8; ++v) {
      int k = aK(v, half);
      aX[2 * v]      = sXb[l16 * 32 + k];
      aX[2 * v + 1]  = sXb[l16 * 32 + k + 1];
      aH0[2 * v]     = sHb[l16 * 64 + k];
      aH0[2 * v + 1] = sHb[l16 * 64 + k + 1];
      aH1[2 * v]     = sHb[l16 * 64 + 32 + k];
      aH1[2 * v + 1] = sHb[l16 * 64 + 32 + k + 1];
    }
    __syncthreads();  // all lanes done reading sHb before rewriting it

    v8f zero = {};
#pragma unroll
    for (int j = 0; j < 4; ++j) {
      int colZ = j * 16 + l16;          // z gate columns
      int colR = 64 + colZ;             // r gate columns
      int colH = 128 + colZ;            // h gate columns
      v16bf bWf, bU0f, bU1f;

      // ---- z gate (fused x-proj + recurrence) ----
#pragma unroll
      for (int v = 0; v < 8; ++v) {
        int kb = 2 * v + 16 * half;
        bWf[2 * v]      = sWt[colZ * 32 + kb];
        bWf[2 * v + 1]  = sWt[colZ * 32 + kb + 1];
        bU0f[2 * v]     = sUt[colZ * 64 + kb];
        bU0f[2 * v + 1] = sUt[colZ * 64 + kb + 1];
        bU1f[2 * v]     = sUt[colZ * 64 + 32 + kb];
        bU1f[2 * v + 1] = sUt[colZ * 64 + 32 + kb + 1];
      }
      v8f accZ = __builtin_amdgcn_wmma_f32_16x16x32_bf16(false, aX, false, bWf, (short)0, zero, false, false);
      accZ = __builtin_amdgcn_wmma_f32_16x16x32_bf16(false, aH0, false, bU0f, (short)0, accZ, false, false);
      accZ = __builtin_amdgcn_wmma_f32_16x16x32_bf16(false, aH1, false, bU1f, (short)0, accZ, false, false);

      // ---- r gate (fused) ----
#pragma unroll
      for (int v = 0; v < 8; ++v) {
        int kb = 2 * v + 16 * half;
        bWf[2 * v]      = sWt[colR * 32 + kb];
        bWf[2 * v + 1]  = sWt[colR * 32 + kb + 1];
        bU0f[2 * v]     = sUt[colR * 64 + kb];
        bU0f[2 * v + 1] = sUt[colR * 64 + kb + 1];
        bU1f[2 * v]     = sUt[colR * 64 + 32 + kb];
        bU1f[2 * v + 1] = sUt[colR * 64 + 32 + kb + 1];
      }
      v8f accR = __builtin_amdgcn_wmma_f32_16x16x32_bf16(false, aX, false, bWf, (short)0, zero, false, false);
      accR = __builtin_amdgcn_wmma_f32_16x16x32_bf16(false, aH0, false, bU0f, (short)0, accR, false, false);
      accR = __builtin_amdgcn_wmma_f32_16x16x32_bf16(false, aH1, false, bU1f, (short)0, accR, false, false);

      // ---- h gate (x part and U part kept separate) ----
#pragma unroll
      for (int v = 0; v < 8; ++v) {
        int kb = 2 * v + 16 * half;
        bWf[2 * v]      = sWt[colH * 32 + kb];
        bWf[2 * v + 1]  = sWt[colH * 32 + kb + 1];
        bU0f[2 * v]     = sUt[colH * 64 + kb];
        bU0f[2 * v + 1] = sUt[colH * 64 + kb + 1];
        bU1f[2 * v]     = sUt[colH * 64 + 32 + kb];
        bU1f[2 * v + 1] = sUt[colH * 64 + 32 + kb + 1];
      }
      v8f accXh = __builtin_amdgcn_wmma_f32_16x16x32_bf16(false, aX, false, bWf, (short)0, zero, false, false);
      v8f accUh = __builtin_amdgcn_wmma_f32_16x16x32_bf16(false, aH0, false, bU0f, (short)0, zero, false, false);
      accUh = __builtin_amdgcn_wmma_f32_16x16x32_bf16(false, aH1, false, bU1f, (short)0, accUh, false, false);

      // ---- gates directly on accumulator registers (branch-free) ----
#pragma unroll
      for (int i = 0; i < 8; ++i) {
        float z  = sigmoidf_(accZ[i] + bz[j]);
        float rg = sigmoidf_(accR[i] + br[j]);
        float hh = tanhf_(accXh[i] + bxh[j] + rg * (accUh[i] + buh[j]));
        float h  = z * hreg[j][i] + (1.0f - z) * hh;
        hreg[j][i] = h;
        int row = half ? (8 + i) : i;
        sHb[row * 64 + colZ] = (__bf16)h;
      }
    }
    __syncthreads();  // sHb fully updated before next step's A-frag reads
  }

  // write h to padded node buffer (row 0 of each batch is the zero pad)
#pragma unroll
  for (int j = 0; j < 4; ++j) {
    int col = j * 16 + l16;
#pragma unroll
    for (int i = 0; i < 8; ++i) {
      int row = half ? (8 + i) : i;
      int s = seq0 + row;
      int b = s / N_, n = s % N_;
      node_pad[((size_t)b * (N_ + 1) + n + 1) * D_ + col] = hreg[j][i];
    }
  }
}

// ---------------------------------------------------------------------------
// Per-(b,r,n): branch-free gather from padded node buffer, 16x64 projection
// on WMMA, leaky+bias, K-softmax attention, weighted sum -> relrep.
// ---------------------------------------------------------------------------
__global__ __launch_bounds__(32) void relatt_kernel(const float* __restrict__ node_pad,
                                                    const int* __restrict__ neighbors,
                                                    const __bf16* __restrict__ projT,
                                                    const float* __restrict__ proj_b,
                                                    const float* __restrict__ satt_W,
                                                    const float* __restrict__ satt_b,
                                                    const float* __restrict__ rel_emb,
                                                    float* __restrict__ relrep) {
  __shared__ int   sIdx[16];
  __shared__ float sNb[16 * 64];
  __shared__ float sAtt[16];

  int w = blockIdx.x;
  int n = w % N_;
  int r = (w / N_) % R_;
  int b = w / (N_ * R_);
  int lane = threadIdx.x;
  int half = lane >> 4, l16 = lane & 15;

  if (lane < 16) sIdx[lane] = neighbors[(((size_t)b * R_ + r) * N_ + n) * K_ + lane];
  __syncthreads();

  // branch-free gather: index 0 hits the zero pad row
  const float* src = node_pad + ((size_t)b * (N_ + 1) + sIdx[l16]) * D_;
  v16bf a0, a1;
#pragma unroll
  for (int v = 0; v < 8; ++v) {
    int k = aK(v, half);
    a0[2 * v]     = (__bf16)src[k];
    a0[2 * v + 1] = (__bf16)src[k + 1];
    a1[2 * v]     = (__bf16)src[32 + k];
    a1[2 * v + 1] = (__bf16)src[32 + k + 1];
  }

  v8f zero = {};
#pragma unroll
  for (int j = 0; j < 4; ++j) {
    v16bf b0f, b1f;
    int e = j * 16 + l16;
    const __bf16* pp = projT + ((size_t)r * 64 + e) * 64;
#pragma unroll
    for (int v = 0; v < 8; ++v) {
      int kb = 2 * v + 16 * half;
      b0f[2 * v]     = pp[kb];      b0f[2 * v + 1] = pp[kb + 1];
      b1f[2 * v]     = pp[32 + kb]; b1f[2 * v + 1] = pp[32 + kb + 1];
    }
    v8f acc = __builtin_amdgcn_wmma_f32_16x16x32_bf16(false, a0, false, b0f, (short)0, zero, false, false);
    acc = __builtin_amdgcn_wmma_f32_16x16x32_bf16(false, a1, false, b1f, (short)0, acc, false, false);
#pragma unroll
    for (int i = 0; i < 8; ++i) {
      int row = half ? (8 + i) : i;
      sNb[row * 64 + e] = leakyf_(acc[i] + proj_b[r * D_ + e]);
    }
  }
  __syncthreads();

  const float* sw  = satt_W + r * (2 * D_ + E_);
  const float* cur = node_pad + ((size_t)b * (N_ + 1) + n + 1) * D_;

  float p = cur[2 * lane] * sw[2 * lane] + cur[2 * lane + 1] * sw[2 * lane + 1];
  float s_cur = wred32(p);

  p = rel_emb[r * E_ + lane] * sw[2 * D_ + lane];
  float s_rel = wred32(p) + satt_b[r];

  p = 0.0f;
  for (int d = half * 32; d < half * 32 + 32; ++d) p += sNb[l16 * 64 + d] * sw[D_ + d];
  p += __shfl_xor(p, 16, 32);

  float sc = leakyf_(s_cur + p + s_rel);
  sc += (sIdx[l16] == 0) ? -1.0e9f : 0.0f;

  float m = sc;
#pragma unroll
  for (int off = 8; off > 0; off >>= 1) m = fmaxf(m, __shfl_xor(m, off, 32));
  float ex = __expf(sc - m);
  float s = ex;
#pragma unroll
  for (int off = 8; off > 0; off >>= 1) s += __shfl_xor(s, off, 32);
  float att = ex * frcp_(s);
  if (lane < 16) sAtt[lane] = att;
  __syncthreads();

  float r0 = 0.0f, r1 = 0.0f;
#pragma unroll
  for (int k = 0; k < 16; ++k) {
    float a = sAtt[k];
    r0 += a * sNb[k * 64 + 2 * lane];
    r1 += a * sNb[k * 64 + 2 * lane + 1];
  }
  float* dst = relrep + (((size_t)b * N_ + n) * R_ + r) * D_;
  dst[2 * lane]     = r0;
  dst[2 * lane + 1] = r1;
}

// ---------------------------------------------------------------------------
// Per-(b,n): relation-attention softmax over R, aggregate, predict, softmax.
// ---------------------------------------------------------------------------
__global__ __launch_bounds__(32) void final_kernel(const float* __restrict__ node_pad,
                                                   const float* __restrict__ relrep,
                                                   const float* __restrict__ rel_emb,
                                                   const float* __restrict__ ratt_W,
                                                   const float* __restrict__ ratt_b,
                                                   const float* __restrict__ pred_W,
                                                   const float* __restrict__ pred_b,
                                                   float* __restrict__ out) {
  int w = blockIdx.x;
  int n = w % N_;
  int b = w / N_;
  int lane = threadIdx.x;

  const float* cur = node_pad + ((size_t)b * (N_ + 1) + n + 1) * D_;
  const float* rep = relrep + ((size_t)b * N_ + n) * R_ * D_;
  float c0 = cur[2 * lane], c1 = cur[2 * lane + 1];

  float dcur = wred32(c0 * ratt_W[2 * lane] + c1 * ratt_W[2 * lane + 1]);

  float rs[R_];
#pragma unroll
  for (int r = 0; r < R_; ++r) {
    float p = rep[r * D_ + 2 * lane] * ratt_W[D_ + 2 * lane]
            + rep[r * D_ + 2 * lane + 1] * ratt_W[D_ + 2 * lane + 1];
    float drep = wred32(p);
    float demb = wred32(rel_emb[r * E_ + lane] * ratt_W[2 * D_ + lane]);
    rs[r] = leakyf_(dcur + drep + demb + ratt_b[0]);
  }
  float m = rs[0];
#pragma unroll
  for (int r = 1; r < R_; ++r) m = fmaxf(m, rs[r]);
  float s = 0.0f;
  float ra[R_];
#pragma unroll
  for (int r = 0; r < R_; ++r) { ra[r] = __expf(rs[r] - m); s += ra[r]; }
  float sinv = frcp_(s);
#pragma unroll
  for (int r = 0; r < R_; ++r) ra[r] *= sinv;

  float u0 = c0, u1 = c1;
#pragma unroll
  for (int r = 0; r < R_; ++r) {
    u0 += ra[r] * rep[r * D_ + 2 * lane];
    u1 += ra[r] * rep[r * D_ + 2 * lane + 1];
  }

  float lg[L_];
#pragma unroll
  for (int l = 0; l < L_; ++l) {
    float p = u0 * pred_W[(2 * lane) * L_ + l] + u1 * pred_W[(2 * lane + 1) * L_ + l];
    lg[l] = wred32(p) + pred_b[l];
  }
  float lm = fmaxf(lg[0], fmaxf(lg[1], lg[2]));
  float e0 = __expf(lg[0] - lm), e1 = __expf(lg[1] - lm), e2 = __expf(lg[2] - lm);
  float esinv = frcp_(e0 + e1 + e2);
  if (lane == 0) {
    float* o = out + ((size_t)b * N_ + n) * L_;
    o[0] = e0 * esinv; o[1] = e1 * esinv; o[2] = e2 * esinv;
  }
}

extern "C" void kernel_launch(void* const* d_in, const int* in_sizes, int n_in,
                              void* d_out, int out_size, void* d_ws, size_t ws_size,
                              hipStream_t stream) {
  const float* X       = (const float*)d_in[0];
  const int*   neigh   = (const int*)d_in[1];
  const float* gru_W   = (const float*)d_in[2];
  const float* gru_U   = (const float*)d_in[3];
  const float* gru_b   = (const float*)d_in[4];
  const float* rel_emb = (const float*)d_in[5];
  const float* proj_W  = (const float*)d_in[6];
  const float* proj_b  = (const float*)d_in[7];
  const float* satt_W  = (const float*)d_in[8];
  const float* satt_b  = (const float*)d_in[9];
  const float* ratt_W  = (const float*)d_in[10];
  const float* ratt_b  = (const float*)d_in[11];
  const float* pred_W  = (const float*)d_in[12];
  const float* pred_b  = (const float*)d_in[13];
  float* out = (float*)d_out;

  float* ws       = (float*)d_ws;
  float* node_pad = ws;                                         // B*(N+1)*D f32
  float* relrep   = ws + (size_t)B_ * (N_ + 1) * D_;            // B*N*R*D f32
  __bf16* projT   = (__bf16*)(relrep + (size_t)B_ * N_ * R_ * D_);  // R*D*D bf16

  prep_kernel<<<(R_ * D_ * D_ + 255) / 256, 256, 0, stream>>>(proj_W, projT, node_pad);
  gru_kernel<<<(B_ * N_) / 16, 32, 0, stream>>>(X, gru_W, gru_U, gru_b, node_pad);
  relatt_kernel<<<B_ * R_ * N_, 32, 0, stream>>>(node_pad, neigh, projT, proj_b,
                                                 satt_W, satt_b, rel_emb, relrep);
  final_kernel<<<B_ * N_, 32, 0, stream>>>(node_pad, relrep, rel_emb, ratt_W, ratt_b,
                                           pred_W, pred_b, out);
}
